// RowParallelLinearWithPacked_28973849379120
// MI455X (gfx1250) — matrix operand order repaired and checked by reading, not compiled, not run on hardware
//
#include <hip/hip_runtime.h>
#include <stdint.h>

typedef __attribute__((ext_vector_type(16))) __bf16    v16bf;
typedef __attribute__((ext_vector_type(8)))  float     v8f;
typedef __attribute__((ext_vector_type(4)))  uint32_t  u32x4;
typedef __attribute__((ext_vector_type(4)))  float     f32x4;

#define NUM_TOKENS  4096
#define K_DIM       4096
#define N_DIM       4096
#define NUM_EXPERTS 4
#define BM 128
#define BN 128
#define BK 32
#define ROW_TILES ((NUM_TOKENS + BM - 1) / BM)   // 32 (worst case: all tokens one expert)
#define COL_TILES (N_DIM / BN)                   // 32
#define LSTRIDE 40                               // bf16 elems per LDS row (32 + 8 pad, keeps 16B align)

// Split one fp32 pair into packed bf16-hi and bf16-lo dwords (RNE both stages).
__device__ __forceinline__ void split2(float f0, float f1, uint32_t& hi, uint32_t& lo) {
  uint32_t a0 = __float_as_uint(f0), a1 = __float_as_uint(f1);
  uint32_t h0 = (a0 + 0x7FFFu + ((a0 >> 16) & 1u)) & 0xFFFF0000u;
  uint32_t h1 = (a1 + 0x7FFFu + ((a1 >> 16) & 1u)) & 0xFFFF0000u;
  float r0 = f0 - __uint_as_float(h0);
  float r1 = f1 - __uint_as_float(h1);
  uint32_t b0 = __float_as_uint(r0), b1 = __float_as_uint(r1);
  uint32_t l0 = b0 + 0x7FFFu + ((b0 >> 16) & 1u);
  uint32_t l1 = b1 + 0x7FFFu + ((b1 >> 16) & 1u);
  hi = (h0 >> 16) | h1;
  lo = (l0 >> 16) | (l1 & 0xFFFF0000u);
}

__global__ void zero_counts_kernel(int* counts) {
  if (threadIdx.x < NUM_EXPERTS) counts[threadIdx.x] = 0;
}

__global__ void build_lists_kernel(const int* __restrict__ indices,
                                   int* __restrict__ counts,
                                   int* __restrict__ lists) {
  int t = blockIdx.x * blockDim.x + threadIdx.x;
  if (t < NUM_TOKENS) {
    int e = indices[t] & (NUM_EXPERTS - 1);
    int p = atomicAdd(&counts[e], 1);
    lists[e * NUM_TOKENS + p] = t;
  }
}

__global__ __launch_bounds__(256)
void grouped_gemm_bf16x3_kernel(const float* __restrict__ x,
                                const float* __restrict__ w,
                                const int* __restrict__ counts,
                                const int* __restrict__ lists,
                                float* __restrict__ out) {
  // [buf][hi=0/lo=1][row*LSTRIDE + k]
  __shared__ __align__(16) __bf16 sA[2][2][BM * LSTRIDE];
  __shared__ __align__(16) __bf16 sB[2][2][BN * LSTRIDE];
  __shared__ int sTok[BM];

  const int b   = blockIdx.x;
  const int e   = b / (ROW_TILES * COL_TILES);
  const int rem = b % (ROW_TILES * COL_TILES);
  const int rt  = rem / COL_TILES;
  const int ct  = rem % COL_TILES;
  const int cnt = counts[e];
  if (rt * BM >= cnt) return;

  const int tid = threadIdx.x;
  if (tid < BM) {
    int row = rt * BM + tid;
    sTok[tid] = (row < cnt) ? lists[e * NUM_TOKENS + row] : -1;
  }
  __syncthreads();

  // ---- global->LDS staging mapping: thread covers (row = tid&127, 16 k-elems) ----
  const int lrow = tid & (BM - 1);
  const int lseg = tid >> 7;                       // 0..1 -> k offset 0 / 16
  int tokRaw = sTok[lrow];
  int tok = (tokRaw < 0) ? sTok[0] : tokRaw;       // clamp padding rows to a valid row
  const float* aG = x + (size_t)tok * K_DIM + lseg * 16;
  const float* bG = w + (size_t)e * N_DIM * K_DIM
                      + (size_t)(ct * BN + lrow) * K_DIM + lseg * 16;

  // ---- wave mapping: 8 waves, 2 (M) x 4 (N); wave tile 64x32 = 4x2 WMMA tiles ----
  const int lane = tid & 31;
  const int wave = tid >> 5;
  const int wm   = wave >> 2;   // 0..1
  const int wn   = wave & 3;    // 0..3
  const int half = lane >> 4;   // 0..1
  const int lm   = lane & 15;

  v8f acc[4][2];
#pragma unroll
  for (int mt = 0; mt < 4; ++mt)
#pragma unroll
    for (int nt = 0; nt < 2; ++nt)
#pragma unroll
      for (int q = 0; q < 8; ++q) acc[mt][nt][q] = 0.0f;

  f32x4 ra[4], rb[4];

  auto loadG = [&](int k0) {
    const f32x4* ap = (const f32x4*)(aG + k0);
    const f32x4* bp = (const f32x4*)(bG + k0);
#pragma unroll
    for (int i = 0; i < 4; ++i) { ra[i] = ap[i]; rb[i] = bp[i]; }
  };

  auto storeL = [&](int buf) {
    uint32_t hA[8], lA[8], hB[8], lB[8];
#pragma unroll
    for (int i = 0; i < 4; ++i) {
      split2(ra[i].x, ra[i].y, hA[2*i],   lA[2*i]);
      split2(ra[i].z, ra[i].w, hA[2*i+1], lA[2*i+1]);
      split2(rb[i].x, rb[i].y, hB[2*i],   lB[2*i]);
      split2(rb[i].z, rb[i].w, hB[2*i+1], lB[2*i+1]);
    }
    const int off = lrow * LSTRIDE + lseg * 16;
    u32x4* p;
    u32x4 v0, v1;
    v0 = (u32x4){hA[0], hA[1], hA[2], hA[3]}; v1 = (u32x4){hA[4], hA[5], hA[6], hA[7]};
    p = (u32x4*)&sA[buf][0][off]; p[0] = v0; p[1] = v1;
    v0 = (u32x4){lA[0], lA[1], lA[2], lA[3]}; v1 = (u32x4){lA[4], lA[5], lA[6], lA[7]};
    p = (u32x4*)&sA[buf][1][off]; p[0] = v0; p[1] = v1;
    v0 = (u32x4){hB[0], hB[1], hB[2], hB[3]}; v1 = (u32x4){hB[4], hB[5], hB[6], hB[7]};
    p = (u32x4*)&sB[buf][0][off]; p[0] = v0; p[1] = v1;
    v0 = (u32x4){lB[0], lB[1], lB[2], lB[3]}; v1 = (u32x4){lB[4], lB[5], lB[6], lB[7]};
    p = (u32x4*)&sB[buf][1][off]; p[0] = v0; p[1] = v1;
  };

  // A fragment (16x32 bf16): lane half selects K chunks {half*8..+8} and {16+half*8..+8}
  auto loadFragA = [&](const __bf16* base, int row) -> v16bf {
    union { v16bf v; u32x4 q[2]; } f;
    const __bf16* p = base + row * LSTRIDE + half * 8;
    f.q[0] = *(const u32x4*)(p);
    f.q[1] = *(const u32x4*)(p + 16);
    return f.v;
  };
  // B fragment (32x16 bf16): lane half selects K chunks {half*8..+8} and {16+half*8..+8}
  auto loadFragB = [&](const __bf16* base, int row) -> v16bf {
    union { v16bf v; u32x4 q[2]; } f;
    const __bf16* p = base + row * LSTRIDE + half * 8;
    f.q[0] = *(const u32x4*)(p);
    f.q[1] = *(const u32x4*)(p + 16);
    return f.v;
  };

  auto compute = [&](int buf) {
    v16bf aH[4], aL[4], bH[2], bL[2];
#pragma unroll
    for (int mt = 0; mt < 4; ++mt) {
      int row = wm * 64 + mt * 16 + lm;
      aH[mt] = loadFragA(sA[buf][0], row);
      aL[mt] = loadFragA(sA[buf][1], row);
    }
#pragma unroll
    for (int nt = 0; nt < 2; ++nt) {
      int row = wn * 32 + nt * 16 + lm;
      bH[nt] = loadFragB(sB[buf][0], row);
      bL[nt] = loadFragB(sB[buf][1], row);
    }
#pragma unroll
    for (int mt = 0; mt < 4; ++mt)
#pragma unroll
      for (int nt = 0; nt < 2; ++nt) {
        acc[mt][nt] = __builtin_amdgcn_wmma_f32_16x16x32_bf16(
            false, aH[mt], false, bH[nt], (short)0, acc[mt][nt], false, false);
        acc[mt][nt] = __builtin_amdgcn_wmma_f32_16x16x32_bf16(
            false, aH[mt], false, bL[nt], (short)0, acc[mt][nt], false, false);
        acc[mt][nt] = __builtin_amdgcn_wmma_f32_16x16x32_bf16(
            false, aL[mt], false, bH[nt], (short)0, acc[mt][nt], false, false);
      }
  };

  // ---- main K loop, double-buffered LDS, one barrier per iteration ----
  loadG(0);
  storeL(0);
  __syncthreads();
  int buf = 0;
  for (int k0 = 0; k0 < K_DIM; k0 += BK) {
    const bool more = (k0 + BK) < K_DIM;
    if (more) loadG(k0 + BK);
    compute(buf);
    if (more) storeL(buf ^ 1);
    __syncthreads();
    buf ^= 1;
  }

  // ---- epilogue: scatter rows through token list; C layout lane=N, vgpr j = M ----
  const int colBase = ct * BN + wn * 32 + lm;
#pragma unroll
  for (int mt = 0; mt < 4; ++mt) {
#pragma unroll
    for (int j = 0; j < 8; ++j) {
      int rloc = wm * 64 + mt * 16 + half * 8 + j;
      int tokr = sTok[rloc];
      if (tokr >= 0) {
        float* op = out + (size_t)tokr * N_DIM + colBase;
        op[0]  = acc[mt][0][j];
        op[16] = acc[mt][1][j];
      }
    }
  }
}

extern "C" void kernel_launch(void* const* d_in, const int* in_sizes, int n_in,
                              void* d_out, int out_size, void* d_ws, size_t ws_size,
                              hipStream_t stream) {
  const float* x       = (const float*)d_in[0];
  const float* w       = (const float*)d_in[1];
  const int*   indices = (const int*)d_in[2];
  float* out = (float*)d_out;

  int* counts = (int*)d_ws;       // 8 ints (4 used)
  int* lists  = counts + 8;       // 4 * 4096 ints  (~64KB workspace total)

  zero_counts_kernel<<<1, 32, 0, stream>>>(counts);
  build_lists_kernel<<<NUM_TOKENS / 256, 256, 0, stream>>>(indices, counts, lists);
  grouped_gemm_bf16x3_kernel<<<NUM_EXPERTS * ROW_TILES * COL_TILES, 256, 0, stream>>>(
      x, w, counts, lists, out);
}
